// Qwen3NextAttention_54357106098373
// MI455X (gfx1250) — matrix operand order repaired and checked by reading, not compiled, not run on hardware
//
#include <hip/hip_runtime.h>
#include <hip/hip_bf16.h>

// ---------------------------------------------------------------------------
// Qwen3-Next attention block for MI455X (gfx1250, wave32, WMMA).
// bf16 operands + f32 accumulation via v_wmma_f32_16x16x32_bf16;
// RMSNorm / RoPE / softmax / sigmoid in fp32.
// CDNA5 features: WMMA, TDM tensor_load_to_lds (+ s_wait_tensorcnt),
// global_prefetch_b8, wave32 shuffles, LDS double buffering.
// ---------------------------------------------------------------------------

#define SEQ    2048
#define HID    2048
#define HD     256
#define NH     16
#define NKV    2
#define QKV_N  9216       // (NH*2 + 2*NKV) * HD
#define QDIM   8192       // NH*2*HD

typedef __attribute__((ext_vector_type(16))) __bf16 v16bf;
typedef __attribute__((ext_vector_type(8)))  __bf16 v8bf;
typedef __attribute__((ext_vector_type(4)))  __bf16 v4bf;
typedef __attribute__((ext_vector_type(8)))  float  v8f;
typedef __attribute__((ext_vector_type(4)))  unsigned int u32x4;
typedef __attribute__((ext_vector_type(8)))  int  i32x8;
typedef __attribute__((ext_vector_type(4)))  int  i32x4;

union Frag16 { v16bf v; v8bf h[2]; __bf16 e[16]; };
union B4     { v4bf  v; __bf16 e[4]; };

#if __has_builtin(__builtin_amdgcn_tensor_load_to_lds)
#define HAS_TDM 1
#else
#define HAS_TDM 0
#endif

// ---------------------------------------------------------------------------
// GEMM:  C[M,N] = A[M,K] * B[N,K]^T     (A,B fp32 row-major; C fp32)
// 128x128 block tile, K-step 32, double-buffered LDS.  8 waves in 2(M)x4(N);
// each wave owns 64x32 = 4x2 WMMA accumulators.  fp32->bf16 conversion while
// staging to LDS; next tile's global loads issue before the WMMA block.
// ---------------------------------------------------------------------------
#define TM 128
#define TN 128
#define TK 32

__global__ __launch_bounds__(256) void gemm_nt(
    const float* __restrict__ A, const float* __restrict__ B,
    float* __restrict__ C, int M, int N, int K)
{
    __shared__ __bf16 As[2][TM * TK];   // [m][k]
    __shared__ __bf16 Bs[2][TN * TK];   // [n][k]

    const int tid  = threadIdx.x;
    const int lane = tid & 31;
    const int wave = tid >> 5;
    const int wm   = wave >> 2;          // 0..1
    const int wn   = wave & 3;           // 0..3
    const int l15  = lane & 15;
    const int lhi  = lane >> 4;

    const int mBlock = blockIdx.y * TM;
    const int nBlock = blockIdx.x * TN;

    const int srow = tid >> 3;           // 0..31 staging row
    const int scg  = tid & 7;            // 0..7  staging col-group (x4 floats)

    v8f acc[4][2] = {};
    float4 fa[4], fb[4];                 // in-flight staging registers

    auto loadTiles = [&](int k0) {
        for (int r = 0; r < 4; ++r) {
            fa[r] = *(const float4*)(A + (size_t)(mBlock + srow + r * 32) * K + k0 + scg * 4);
            fb[r] = *(const float4*)(B + (size_t)(nBlock + srow + r * 32) * K + k0 + scg * 4);
        }
    };
    auto storeTiles = [&](int buf) {
        for (int r = 0; r < 4; ++r) {
            B4 pa; pa.e[0] = (__bf16)fa[r].x; pa.e[1] = (__bf16)fa[r].y;
                   pa.e[2] = (__bf16)fa[r].z; pa.e[3] = (__bf16)fa[r].w;
            *(v4bf*)(&As[buf][(srow + r * 32) * TK + scg * 4]) = pa.v;
            B4 pb; pb.e[0] = (__bf16)fb[r].x; pb.e[1] = (__bf16)fb[r].y;
                   pb.e[2] = (__bf16)fb[r].z; pb.e[3] = (__bf16)fb[r].w;
            *(v4bf*)(&Bs[buf][(srow + r * 32) * TK + scg * 4]) = pb.v;
        }
    };

    const int nk = K / TK;
    loadTiles(0);
    storeTiles(0);
    __syncthreads();

    for (int kt = 0; kt < nk; ++kt) {
        const int buf = kt & 1;
        if (kt + 1 < nk) loadTiles((kt + 1) * TK);        // overlap with WMMA
        if (kt + 2 < nk) {                                // global_prefetch_b8
            __builtin_prefetch(A + (size_t)(mBlock + srow) * K + (kt + 2) * TK + scg * 4, 0, 0);
            __builtin_prefetch(B + (size_t)(nBlock + srow) * K + (kt + 2) * TK + scg * 4, 0, 0);
        }

        // ---- fragments (CDNA5 wave32 WMMA layouts) ----
        Frag16 afrag[4], bfrag[2];
        for (int i = 0; i < 4; ++i) {
            // A 16x32: elem j -> K = (j>>3)*16 + lhi*8 + (j&7), M = l15
            const __bf16* base = &As[buf][(wm * 64 + i * 16 + l15) * TK + lhi * 8];
            afrag[i].h[0] = *(const v8bf*)(base);
            afrag[i].h[1] = *(const v8bf*)(base + 16);
        }
        for (int j = 0; j < 2; ++j) {
            // B 32x16: elem j -> K = (j>>3)*16 + lhi*8 + (j&7), N = l15 ; Bs stores [n][k]
            const __bf16* base = &Bs[buf][(wn * 32 + j * 16 + l15) * TK + lhi * 8];
            bfrag[j].h[0] = *(const v8bf*)(base);
            bfrag[j].h[1] = *(const v8bf*)(base + 16);
        }
        for (int i = 0; i < 4; ++i)
            for (int j = 0; j < 2; ++j)
                acc[i][j] = __builtin_amdgcn_wmma_f32_16x16x32_bf16(
                    false, afrag[i].v, false, bfrag[j].v,
                    (short)0, acc[i][j], false, false);

        if (kt + 1 < nk) storeTiles(buf ^ 1);             // fill other buffer
        __syncthreads();
    }

    // ---- epilogue: C/D layout -> lane l15 = col, VGPR r = row (r + 8*lhi) ----
    for (int i = 0; i < 4; ++i)
        for (int j = 0; j < 2; ++j) {
            const int mT = mBlock + wm * 64 + i * 16 + lhi * 8;
            const int nT = nBlock + wn * 32 + j * 16 + l15;
            for (int r = 0; r < 8; ++r)
                C[(size_t)(mT + r) * N + nT] = acc[i][j][r];
        }
}

// ---------------------------------------------------------------------------
// RMSNorm + RoPE + bf16 pack for Q and K; bf16 pack for V.
// One block per token; lane L holds d = L*8..L*8+7 so rotate_half's
// d <-> d^128 is lane L ^ 16.  1/sqrt(HD) folded into Q.
// ---------------------------------------------------------------------------
__global__ __launch_bounds__(256) void qkv_post(
    const float* __restrict__ qkv, const float* __restrict__ cs,
    const float* __restrict__ sn,  const float* __restrict__ wq,
    const float* __restrict__ wk,
    __bf16* __restrict__ qb, __bf16* __restrict__ kb, __bf16* __restrict__ vb)
{
    const int t    = blockIdx.x;
    const int lane = threadIdx.x & 31;
    const int wave = threadIdx.x >> 5;
    const float* row = qkv + (size_t)t * QKV_N;
    const float* cr  = cs  + (size_t)t * HD;
    const float* sr  = sn  + (size_t)t * HD;
    const int dbase  = lane * 8;

    for (int u = wave; u < 20; u += 8) {
        if (u < 18) {                       // Q heads (0..15) and K heads (16..17)
            const bool isQ = (u < 16);
            const int  hh  = isQ ? u : (u - 16);
            const float* src = isQ ? (row + u * (2 * HD)) : (row + QDIM + hh * HD);
            const float* w   = isQ ? wq : wk;

            float x[8]; float ss = 0.f;
            for (int i = 0; i < 8; ++i) { x[i] = src[dbase + i]; ss += x[i] * x[i]; }
            for (int off = 16; off; off >>= 1) ss += __shfl_xor(ss, off, 32);
            const float inv    = rsqrtf(ss * (1.0f / HD) + 1e-6f);
            const float sgn    = (lane < 16) ? -1.0f : 1.0f;
            const float oscale = isQ ? 0.0625f : 1.0f;   // 1/sqrt(256) for Q

            B4 o0, o1;
            for (int i = 0; i < 8; ++i) {
                const int d  = dbase + i;
                const float xn = x[i] * inv * w[d];
                const float xr = __shfl_xor(xn, 16, 32);          // norm val at d^128
                float val = xn * cr[d] + sgn * xr * sr[d];
                val *= oscale;
                if (i < 4) o0.e[i] = (__bf16)val; else o1.e[i - 4] = (__bf16)val;
            }
            __bf16* dst = isQ ? (qb + ((size_t)t * NH  + hh) * HD + dbase)
                              : (kb + ((size_t)t * NKV + hh) * HD + dbase);
            *(v4bf*)dst       = o0.v;
            *(v4bf*)(dst + 4) = o1.v;
        } else {                            // V heads (18..19): just convert
            const int hh = u - 18;
            const float* src = row + QDIM + NKV * HD + hh * HD;
            B4 o0, o1;
            for (int i = 0; i < 8; ++i) {
                const float val = src[dbase + i];
                if (i < 4) o0.e[i] = (__bf16)val; else o1.e[i - 4] = (__bf16)val;
            }
            __bf16* dst = vb + ((size_t)t * NKV + hh) * HD + dbase;
            *(v4bf*)dst       = o0.v;
            *(v4bf*)(dst + 4) = o1.v;
        }
    }
}

// ---------------------------------------------------------------------------
// Flash attention (causal, GQA g=8), online softmax in fp32.
// Block = 4 waves = 32 queries; grid = (SEQ/32, NH).
// Wave w: query group (w>>1)*16, dim half (w&1)*128.
// S^T = K * Q^T so the per-query softmax reduction is intra-lane + one
// shfl_xor(16).  The 32x256 bf16 V tile for each key block is DMA'd into LDS
// by the Tensor Data Mover (tensor_load_to_lds + s_wait_tensorcnt), replacing
// strided scalar global gathers with LDS reads.
// ---------------------------------------------------------------------------
__global__ __launch_bounds__(128) void flash_attn(
    const __bf16* __restrict__ qb, const __bf16* __restrict__ kb,
    const __bf16* __restrict__ vb, float* __restrict__ attn)
{
    __shared__ __bf16 Ps[4][16 * 32];     // per-wave P[query][key] bf16 tile
#if HAS_TDM
    __shared__ __bf16 Vs[32 * HD];        // 16KB V tile, [key][dim]
#endif

    const int head = blockIdx.y;
    const int qblk = blockIdx.x;          // block of 32 queries
    const int lane = threadIdx.x & 31;
    const int wave = threadIdx.x >> 5;
    const int qg   = wave >> 1;           // query group 0/1
    const int dh   = wave & 1;            // dim half 0/1
    const int l15  = lane & 15;
    const int lhi  = lane >> 4;
    const int kvh  = head >> 3;           // GQA: 8 q-heads per kv-head
    const int qRow0 = qblk * 32 + qg * 16;
    const int dim0  = dh * 128;

    // Q fragments, B layout (N=query=l15, K=dim=(j>>3)*16+lhi*8+(j&7)), 1/sqrt(HD) pre-folded
    Frag16 qf[8];
    {
        const __bf16* qp = qb + ((size_t)(qRow0 + l15) * NH + head) * HD;
        for (int c = 0; c < 8; ++c) {
            const __bf16* base = qp + c * 32 + lhi * 8;
            qf[c].h[0] = *(const v8bf*)(base);
            qf[c].h[1] = *(const v8bf*)(base + 16);
        }
    }

    v8f acc[8] = {};                      // 16 queries x 128 dims (fp32)
    float m_run = -3.0e38f, l_run = 0.0f;

    for (int kb2 = 0; kb2 <= qblk; ++kb2) {
        const int kBase = kb2 * 32;

#if HAS_TDM
        // ---- TDM: DMA V[kBase..kBase+31][0..255] (bf16) into LDS ----
        __syncthreads();                  // previous Vs fully consumed
        if (wave == 0) {
            const unsigned long long gaddr =
                (unsigned long long)(vb + ((size_t)kBase * NKV + kvh) * HD);
            const unsigned ldsoff = (unsigned)(size_t)(&Vs[0]);
            u32x4 g0;
            g0[0] = 1u;                                   // count=1 valid D#
            g0[1] = ldsoff;                               // lds_addr (bytes)
            g0[2] = (unsigned)(gaddr & 0xffffffffu);      // global_addr[31:0]
            g0[3] = (unsigned)((gaddr >> 32) & 0x01ffffffu) | 0x80000000u; // type=2
            i32x8 g1;
            g1[0] = (int)(1u << 16);                      // data_size=2B, no mask
            g1[1] = (int)((unsigned)(HD & 0xffff) << 16); // tensor_dim0 = 256
            g1[2] = (int)((unsigned)((SEQ * NKV) & 0xffff) << 16); // tensor_dim1
            g1[3] = (int)((unsigned)(HD & 0xffff) << 16); // tile_dim0 = 256
            g1[4] = 32;                                   // tile_dim1 = 32 keys
            g1[5] = NKV * HD;                             // tensor_dim0_stride = 512
            g1[6] = 0;
            g1[7] = 0;
            i32x4 gz4; gz4[0] = 0; gz4[1] = 0; gz4[2] = 0; gz4[3] = 0;
            i32x8 gz8; gz8[0] = 0; gz8[1] = 0; gz8[2] = 0; gz8[3] = 0;
                       gz8[4] = 0; gz8[5] = 0; gz8[6] = 0; gz8[7] = 0;
            // clang-23 / amdgpu-toolchain 6-arg form:
            // (u32x4 g0, i32x8 g1, i32x4 g2, i32x4 g3, i32x8 extra, i32 cpol)
            __builtin_amdgcn_tensor_load_to_lds(g0, g1, gz4, gz4, gz8, 0);
            __builtin_amdgcn_s_wait_tensorcnt(0);
        }
        __syncthreads();                  // publish Vs to all waves
#endif

        // ---- S^T (keys x queries): two 16x16 tiles, K-dim = HD in 8 chunks ----
        v8f st[2] = {};
        for (int c = 0; c < 8; ++c)
            for (int kt = 0; kt < 2; ++kt) {
                Frag16 kf;   // A layout: M=key=l15, K=dim
                const __bf16* base = kb +
                    ((size_t)(kBase + kt * 16 + l15) * NKV + kvh) * HD + c * 32 + lhi * 8;
                kf.h[0] = *(const v8bf*)(base);
                kf.h[1] = *(const v8bf*)(base + 16);
                st[kt] = __builtin_amdgcn_wmma_f32_16x16x32_bf16(
                    false, kf.v, false, qf[c].v, (short)0, st[kt], false, false);
            }

        // ---- causal mask on diagonal block ----
        if (kb2 == qblk) {
            const int qRel = qg * 16 + l15;
            for (int kt = 0; kt < 2; ++kt)
                for (int r = 0; r < 8; ++r) {
                    const int kRel = kt * 16 + r + 8 * lhi;
                    if (kRel > qRel) st[kt][r] = -3.0e38f;
                }
        }

        // ---- online softmax (stats for query l15, mirrored in lane^16) ----
        float mloc = st[0][0];
        for (int kt = 0; kt < 2; ++kt)
            for (int r = 0; r < 8; ++r) mloc = fmaxf(mloc, st[kt][r]);
        mloc = fmaxf(mloc, __shfl_xor(mloc, 16, 32));
        const float m_new = fmaxf(m_run, mloc);
        const float alpha = __expf(m_run - m_new);

        float lloc = 0.0f;
        for (int kt = 0; kt < 2; ++kt)
            for (int r = 0; r < 8; ++r) {
                const float p = __expf(st[kt][r] - m_new);
                lloc += p;
                Ps[wave][l15 * 32 + kt * 16 + r + 8 * lhi] = (__bf16)p;
            }
        lloc += __shfl_xor(lloc, 16, 32);
        l_run = l_run * alpha + lloc;
        m_run = m_new;

        // ---- rescale O: per-row alpha gathered via lane shuffle ----
        float alphaM[8];
        for (int r = 0; r < 8; ++r) alphaM[r] = __shfl(alpha, r + 8 * lhi, 32);
        for (int c = 0; c < 8; ++c)
            for (int r = 0; r < 8; ++r) acc[c][r] *= alphaM[r];

        // ---- P as A fragment (M=query=l15, K=key) from LDS ----
        Frag16 pf;
        {
            const __bf16* base = &Ps[wave][l15 * 32 + lhi * 16];
            pf.h[0] = *(const v8bf*)(base);
            pf.h[1] = *(const v8bf*)(base + 8);
        }

        // ---- O += P * V over this wave's 128 dims (8 chunks of 16) ----
        for (int c = 0; c < 8; ++c) {
            Frag16 vf;   // B layout: N=dim=l15, K=key=lhi*16+j
            for (int j = 0; j < 16; ++j) {
#if HAS_TDM
                vf.e[j] = Vs[(lhi * 16 + j) * HD + dim0 + c * 16 + l15];
#else
                const int key = kBase + lhi * 16 + j;
                vf.e[j] = vb[((size_t)key * NKV + kvh) * HD + dim0 + c * 16 + l15];
#endif
            }
            acc[c] = __builtin_amdgcn_wmma_f32_16x16x32_bf16(
                false, pf.v, false, vf.v, (short)0, acc[c], false, false);
        }
    }

    // ---- normalize by l and store fp32 ----
    const float linv = 1.0f / l_run;
    float linvM[8];
    for (int r = 0; r < 8; ++r) linvM[r] = __shfl(linv, r + 8 * lhi, 32);
    for (int c = 0; c < 8; ++c)
        for (int r = 0; r < 8; ++r) {
            const int row = qRow0 + r + 8 * lhi;
            attn[((size_t)row * NH + head) * HD + dim0 + c * 16 + l15] =
                acc[c][r] * linvM[r];
        }
}

// ---------------------------------------------------------------------------
// attn *= sigmoid(gate); gate lives in the qkv buffer at col n*512 + 256 + d.
// ---------------------------------------------------------------------------
__global__ __launch_bounds__(256) void gate_apply(
    const float* __restrict__ qkv, float* __restrict__ attn)
{
    const int idx = blockIdx.x * 256 + threadIdx.x;   // over SEQ*NH*HD
    const int t = idx / (NH * HD);
    const int r = idx % (NH * HD);
    const int n = r / HD;
    const int d = r % HD;
    const float gv = qkv[(size_t)t * QKV_N + n * (2 * HD) + HD + d];
    attn[idx] *= 1.0f / (1.0f + __expf(-gv));
}

// ---------------------------------------------------------------------------
// Launch.  Inputs (setup_inputs order): hidden, cos, sin, Wqkv, Wo, wq, wk.
// ---------------------------------------------------------------------------
extern "C" void kernel_launch(void* const* d_in, const int* in_sizes, int n_in,
                              void* d_out, int out_size, void* d_ws, size_t ws_size,
                              hipStream_t stream)
{
    const float* hidden = (const float*)d_in[0];
    const float* cs     = (const float*)d_in[1];
    const float* sn     = (const float*)d_in[2];
    const float* Wqkv   = (const float*)d_in[3];
    const float* Wo     = (const float*)d_in[4];
    const float* wq     = (const float*)d_in[5];
    const float* wk     = (const float*)d_in[6];
    float* out = (float*)d_out;

    char* ws = (char*)d_ws;
    float*  qkv  = (float*)ws;  ws += (size_t)SEQ * QKV_N * sizeof(float);     // 75.5 MB
    float*  attn = (float*)ws;  ws += (size_t)SEQ * NH * HD * sizeof(float);   // 33.5 MB
    __bf16* qb   = (__bf16*)ws; ws += (size_t)SEQ * NH * HD * sizeof(__bf16);  // 16.8 MB
    __bf16* kb   = (__bf16*)ws; ws += (size_t)SEQ * NKV * HD * sizeof(__bf16); //  2.1 MB
    __bf16* vb   = (__bf16*)ws; ws += (size_t)SEQ * NKV * HD * sizeof(__bf16); //  2.1 MB

    // 1) qkv = hidden @ Wqkv^T        (2048 x 9216, K=2048)
    gemm_nt<<<dim3(QKV_N / TN, SEQ / TM), 256, 0, stream>>>(
        hidden, Wqkv, qkv, SEQ, QKV_N, HID);

    // 2) RMSNorm + RoPE + bf16 pack of Q/K/V
    qkv_post<<<SEQ, 256, 0, stream>>>(qkv, cs, sn, wq, wk, qb, kb, vb);

    // 3) causal flash attention (TDM-fed V tiles)
    flash_attn<<<dim3(SEQ / 32, NH), 128, 0, stream>>>(qb, kb, vb, attn);

    // 4) gated output: attn *= sigmoid(gate)
    gate_apply<<<(SEQ * NH * HD) / 256, 256, 0, stream>>>(qkv, attn);

    // 5) out = gated_attn @ Wo^T      (2048 x 2048, K=4096)
    gemm_nt<<<dim3(HID / TN, SEQ / TM), 256, 0, stream>>>(
        attn, Wo, out, SEQ, HID, NH * HD);
}
